// Deepergcn_dagnn_dist_82351702933566
// MI455X (gfx1250) — matrix operand order repaired and checked by reading, not compiled, not run on hardware
//
#include <hip/hip_runtime.h>
#include <hip/hip_bf16.h>

typedef __attribute__((ext_vector_type(16))) _Float16 v16h;
typedef __attribute__((ext_vector_type(8)))  _Float16 v8h;
typedef __attribute__((ext_vector_type(8)))  float    v8f;

#define NN   1024
#define EE   16384
#define DD   256
#define HD   512
#define LL   4
#define GG   32
#define KHOP 5

// ---------------------------------------------------------------------------
// f32 -> f16 conversion (weights once; activations once per GEMM)
// ---------------------------------------------------------------------------
__global__ __launch_bounds__(256) void k_f32_to_f16(const float* __restrict__ s,
                                                    _Float16* __restrict__ d, int n) {
  int i = blockIdx.x * 256 + threadIdx.x;
  if (i < n) d[i] = (_Float16)s[i];
}

// ---------------------------------------------------------------------------
// WMMA GEMM: Y[M,O] = X[M,K] @ W[O,K]^T + bias[O] (+ res[M,O])
// X,W f16 row-major. One wave computes a 16x64 output tile: one A fragment
// is reused across 4 B fragments -> 4 v_wmma_f32_16x16x32_f16 per K-step.
// ---------------------------------------------------------------------------
__global__ __launch_bounds__(256) void k_gemm_wmma(
    const _Float16* __restrict__ X, const _Float16* __restrict__ W,
    const float* __restrict__ bias, const float* __restrict__ res,
    float* __restrict__ Y, int M, int K, int O) {
  int wave = (blockIdx.x * 256 + (int)threadIdx.x) >> 5;
  int lane = threadIdx.x & 31;
  int otiles = O >> 6;                 // 64-wide O tiles
  int tiles  = (M >> 4) * otiles;
  if (wave >= tiles) return;
  int mt = wave / otiles, ot = wave - mt * otiles;
  int m_base = mt << 4, o_base = ot << 6;
  int hl  = lane >> 4;                 // 0: lanes 0-15, 1: lanes 16-31
  int l15 = lane & 15;

  // A fragment (ISA 16-bit A 16x32): lane holds row m_base+l15;
  // halfs 0..15 = K kb..kb+15 (contiguous, matches B), kb = hl?16:0
  const _Float16* xrow = X + (size_t)(m_base + l15) * K + (hl ? 16 : 0);
  // B fragment (ISA 16-bit B 32x16): lane holds column; 16 contiguous K
  // values at offset koff = hl?16:0
  const _Float16* wrow = W + (size_t)(o_base + l15) * K + (hl ? 16 : 0);
  const size_t wstride = (size_t)16 * K;

  v8f acc0 = {}, acc1 = {}, acc2 = {}, acc3 = {};
  for (int k0 = 0; k0 < K; k0 += 32) {
    v8h alo = *(const v8h*)(xrow + k0);
    v8h ahi = *(const v8h*)(xrow + k0 + 8);
    v16h a;
    for (int t = 0; t < 8; ++t) { a[t] = alo[t]; a[t + 8] = ahi[t]; }
    v16h b0 = *(const v16h*)(wrow + k0);
    v16h b1 = *(const v16h*)(wrow + wstride + k0);
    v16h b2 = *(const v16h*)(wrow + 2 * wstride + k0);
    v16h b3 = *(const v16h*)(wrow + 3 * wstride + k0);
    acc0 = __builtin_amdgcn_wmma_f32_16x16x32_f16(false, a, false, b0, (short)0, acc0, false, false);
    acc1 = __builtin_amdgcn_wmma_f32_16x16x32_f16(false, a, false, b1, (short)0, acc1, false, false);
    acc2 = __builtin_amdgcn_wmma_f32_16x16x32_f16(false, a, false, b2, (short)0, acc2, false, false);
    acc3 = __builtin_amdgcn_wmma_f32_16x16x32_f16(false, a, false, b3, (short)0, acc3, false, false);
  }
  // C/D layout: VGPR r, lanes 0-15 -> (m_base+r, n), lanes 16-31 -> (m_base+8+r, n)
  int mrow = m_base + (hl ? 8 : 0);
  v8f accs[4] = {acc0, acc1, acc2, acc3};
  for (int t = 0; t < 4; ++t) {
    int ocol = o_base + t * 16 + l15;
    float bv = bias[ocol];
    for (int r = 0; r < 8; ++r) {
      size_t idx = (size_t)(mrow + r) * O + ocol;
      float v = accs[t][r] + bv;
      if (res) v += res[idx];
      Y[idx] = v;
    }
  }
}

// ---------------------------------------------------------------------------
// Row-wise LayerNorm (+optional ReLU). One block of 256 threads per row.
// ---------------------------------------------------------------------------
__global__ __launch_bounds__(256) void k_ln_relu(const float* __restrict__ X,
    float* __restrict__ Y, const float* __restrict__ g, const float* __restrict__ b,
    int W, int relu) {
  __shared__ float sbuf[256];
  int row = blockIdx.x, tid = threadIdx.x;
  const float* x = X + (size_t)row * W;
  float* y = Y + (size_t)row * W;
  float s = 0.f;
  for (int i = tid; i < W; i += 256) s += x[i];
  sbuf[tid] = s; __syncthreads();
  for (int o = 128; o > 0; o >>= 1) { if (tid < o) sbuf[tid] += sbuf[tid + o]; __syncthreads(); }
  float mu = sbuf[0] / W; __syncthreads();
  float v = 0.f;
  for (int i = tid; i < W; i += 256) { float d = x[i] - mu; v += d * d; }
  sbuf[tid] = v; __syncthreads();
  for (int o = 128; o > 0; o >>= 1) { if (tid < o) sbuf[tid] += sbuf[tid + o]; __syncthreads(); }
  float rstd = rsqrtf(sbuf[0] / W + 1e-5f);
  for (int i = tid; i < W; i += 256) {
    float z = (x[i] - mu) * rstd * g[i] + b[i];
    y[i] = relu ? fmaxf(z, 0.f) : z;
  }
}

// ---------------------------------------------------------------------------
// Embeddings
// ---------------------------------------------------------------------------
__global__ __launch_bounds__(256) void k_embed_nodes(const float* __restrict__ atom_emb,
    const int* __restrict__ x, const float* __restrict__ vn0,
    const float* __restrict__ rand_h, float* __restrict__ h) {
  int n = blockIdx.x, d = threadIdx.x;
  float acc = vn0[d] + rand_h[(size_t)n * DD + d];
  for (int f = 0; f < 9; ++f) {
    int xv = x[n * 9 + f];
    acc += atom_emb[((size_t)(f * 119 + xv)) * DD + d];
  }
  h[(size_t)n * DD + d] = acc;
}

__global__ __launch_bounds__(256) void k_embed_edges(const float* __restrict__ bond_emb,
    const int* __restrict__ ea, const float* __restrict__ rand_e, float* __restrict__ eattr) {
  int e = blockIdx.x, d = threadIdx.x;
  float acc = rand_e[(size_t)e * DD + d];
  for (int f = 0; f < 3; ++f) {
    int av = ea[e * 3 + f];
    acc += bond_emb[((size_t)(f * 6 + av)) * DD + d];
  }
  eattr[(size_t)e * DD + d] = acc;
}

__global__ __launch_bounds__(256) void k_vn_init(const float* __restrict__ vn0,
                                                 float* __restrict__ vn) {
  vn[(size_t)blockIdx.x * DD + threadIdx.x] = vn0[threadIdx.x];
}

// ---------------------------------------------------------------------------
// Deterministic dst-CSR build (ordered per-node edge scan, no float atomics)
// ---------------------------------------------------------------------------
__global__ __launch_bounds__(256) void k_csr_count(const int* __restrict__ dst,
                                                   int* __restrict__ cnt) {
  int n = blockIdx.x * 256 + threadIdx.x;
  if (n >= NN) return;
  int c = 0;
  for (int e = 0; e < EE; ++e) c += (dst[e] == n);
  cnt[n] = c;
}

__global__ __launch_bounds__(1024) void k_csr_scan(const int* __restrict__ cnt,
                                                   int* __restrict__ off) {
  __shared__ int s[NN];
  int tid = threadIdx.x;
  s[tid] = cnt[tid]; __syncthreads();
  for (int o = 1; o < NN; o <<= 1) {
    int v = (tid >= o) ? s[tid - o] : 0;
    __syncthreads();
    s[tid] += v;
    __syncthreads();
  }
  if (tid == 0) off[0] = 0;
  off[tid + 1] = s[tid];
}

__global__ __launch_bounds__(256) void k_csr_fill(const int* __restrict__ dst,
    const int* __restrict__ off, int* __restrict__ csr_e) {
  int n = blockIdx.x * 256 + threadIdx.x;
  if (n >= NN) return;
  int p = off[n];
  for (int e = 0; e < EE; ++e)
    if (dst[e] == n) csr_e[p++] = e;
}

// ---------------------------------------------------------------------------
// GENConv softmax aggregation: z[n] = hin[n] + softmax-weighted sum of messages
// ---------------------------------------------------------------------------
__global__ __launch_bounds__(256) void k_genconv_aggr(const float* __restrict__ hin,
    const float* __restrict__ eattr, const int* __restrict__ src,
    const int* __restrict__ off, const int* __restrict__ csr_e,
    const float* __restrict__ conv_t, int layer, float* __restrict__ z) {
  int n = blockIdx.x, d = threadIdx.x;
  float t = conv_t[layer];
  int s0 = off[n], s1 = off[n + 1];
  float mx = -__builtin_inff();
  for (int i = s0; i < s1; ++i) {
    int e = csr_e[i], s = src[e];
    float m = fmaxf(hin[(size_t)s * DD + d] + eattr[(size_t)e * DD + d], 0.f) + 1e-7f;
    mx = fmaxf(mx, t * m);
  }
  if (s1 == s0) mx = 0.f;  // isolated node guard (matches isfinite check)
  float den = 0.f, num = 0.f;
  for (int i = s0; i < s1; ++i) {
    int e = csr_e[i], s = src[e];
    float m = fmaxf(hin[(size_t)s * DD + d] + eattr[(size_t)e * DD + d], 0.f) + 1e-7f;
    float ex = expf(t * m - mx);
    den += ex; num += ex * m;
  }
  float aggr = num / fmaxf(den, 1e-16f);
  z[(size_t)n * DD + d] = hin[(size_t)n * DD + d] + aggr;
}

// ---------------------------------------------------------------------------
// Virtual node helpers
// ---------------------------------------------------------------------------
__global__ __launch_bounds__(256) void k_vn_pool(const float* __restrict__ h,
    const float* __restrict__ vn, float* __restrict__ vtmp) {
  int g = blockIdx.x, d = threadIdx.x;
  float acc = vn[(size_t)g * DD + d];
  for (int i = 0; i < NN / GG; ++i) acc += h[(size_t)(g * (NN / GG) + i) * DD + d];
  vtmp[(size_t)g * DD + d] = acc;
}

__global__ __launch_bounds__(256) void k_add_vn(float* __restrict__ h,
    const float* __restrict__ vn, const int* __restrict__ batch) {
  int n = blockIdx.x, d = threadIdx.x;
  h[(size_t)n * DD + d] += vn[(size_t)batch[n] * DD + d];
}

__global__ __launch_bounds__(256) void k_final_combine(const float* __restrict__ t,
    const float* __restrict__ vn, const int* __restrict__ batch,
    float* __restrict__ pps0) {
  int n = blockIdx.x, d = threadIdx.x;
  pps0[(size_t)n * DD + d] = t[(size_t)n * DD + d] + vn[(size_t)batch[n] * DD + d];
}

// ---------------------------------------------------------------------------
// DAGNN
// ---------------------------------------------------------------------------
__global__ __launch_bounds__(256) void k_calc_dis(const int* __restrict__ off,
                                                  float* __restrict__ dis) {
  int n = blockIdx.x * 256 + threadIdx.x;
  if (n >= NN) return;
  dis[n] = rsqrtf((float)(off[n + 1] - off[n] + 1));
}

__global__ __launch_bounds__(256) void k_dagnn_hop(const float* __restrict__ xc,
    float* __restrict__ xn, const float* __restrict__ dis, const int* __restrict__ src,
    const int* __restrict__ off, const int* __restrict__ csr_e) {
  int n = blockIdx.x, d = threadIdx.x;
  float dn = dis[n];
  float acc = dn * dn * xc[(size_t)n * DD + d];
  for (int i = off[n]; i < off[n + 1]; ++i) {
    int e = csr_e[i], s = src[e];
    acc += dis[s] * dn * xc[(size_t)s * DD + d];
  }
  xn[(size_t)n * DD + d] = acc;
}

__global__ __launch_bounds__(256) void k_dagnn_score(const float* __restrict__ pps,
    const float* __restrict__ w, const float* __restrict__ b, float* __restrict__ xs) {
  __shared__ float sbuf[256];
  __shared__ float sc[KHOP + 1];
  int n = blockIdx.x, d = threadIdx.x;
  for (int k = 0; k <= KHOP; ++k) {
    sbuf[d] = pps[((size_t)k * NN + n) * DD + d] * w[d];
    __syncthreads();
    for (int o = 128; o > 0; o >>= 1) { if (d < o) sbuf[d] += sbuf[d + o]; __syncthreads(); }
    if (d == 0) sc[k] = 1.f / (1.f + expf(-(sbuf[0] + b[0])));
    __syncthreads();
  }
  float acc = 0.f;
  for (int k = 0; k <= KHOP; ++k) acc += sc[k] * pps[((size_t)k * NN + n) * DD + d];
  xs[(size_t)n * DD + d] = acc;
}

// ---------------------------------------------------------------------------
// Output: zero-fill out+mask, write mask.sum(), fill block-diagonal pairs
// ---------------------------------------------------------------------------
__global__ __launch_bounds__(256) void k_zero_out(float* __restrict__ out) {
  size_t i = (size_t)blockIdx.x * 256 + threadIdx.x;
  size_t total = 2ull * NN * NN;
  if (i < total) out[i] = 0.f;
  if (i == 0) out[total] = (float)(GG * (NN / GG) * (NN / GG - 1));  // mask.sum()
}

__global__ __launch_bounds__(256) void k_distmax(const float* __restrict__ xs,
    const float* __restrict__ fcw, const float* __restrict__ fcb,
    float* __restrict__ out, float* __restrict__ mask) {
  int g = blockIdx.x >> 2;
  int p = ((blockIdx.x & 3) << 8) + threadIdx.x;   // 0..1023 pairs in graph
  int i = p >> 5, j = p & 31;
  if (i == j) return;
  int I = g * 32 + i, J = g * 32 + j;
  const float* xi = xs + (size_t)I * DD;
  const float* xj = xs + (size_t)J * DD;
  float acc = 0.f;
  for (int d = 0; d < DD; ++d) acc += fmaxf(xi[d], xj[d]) * fcw[d];
  out[(size_t)I * NN + J]  = fmaxf(acc + fcb[0], 0.f);
  mask[(size_t)I * NN + J] = 1.f;
}

// ---------------------------------------------------------------------------
// Host orchestration
// ---------------------------------------------------------------------------
static inline void* bump(char* base, size_t& off, size_t bytes) {
  off = (off + 255) & ~(size_t)255;
  void* p = base + off;
  off += bytes;
  return p;
}

extern "C" void kernel_launch(void* const* d_in, const int* in_sizes, int n_in,
                              void* d_out, int out_size, void* d_ws, size_t ws_size,
                              hipStream_t stream) {
  (void)in_sizes; (void)n_in; (void)out_size; (void)ws_size;
  const float* vn0        = (const float*)d_in[0];
  const float* atom_emb   = (const float*)d_in[1];
  const float* bond_emb   = (const float*)d_in[2];
  const float* rand_h     = (const float*)d_in[3];
  const float* rand_e     = (const float*)d_in[4];
  const float* conv_t     = (const float*)d_in[5];
  const float* w_in       = (const float*)d_in[6];
  const float* b_in       = (const float*)d_in[7];
  const float* w_hid      = (const float*)d_in[8];
  const float* b_hid      = (const float*)d_in[9];
  const float* w_out      = (const float*)d_in[10];
  const float* b_out      = (const float*)d_in[11];
  const float* mlp_ln_g   = (const float*)d_in[12];
  const float* mlp_ln_b   = (const float*)d_in[13];
  const float* layer_ln_g = (const float*)d_in[14];
  const float* layer_ln_b = (const float*)d_in[15];
  const float* vn_w1      = (const float*)d_in[16];
  const float* vn_b1      = (const float*)d_in[17];
  const float* vn_ln1_g   = (const float*)d_in[18];
  const float* vn_ln1_b   = (const float*)d_in[19];
  const float* vn_w2      = (const float*)d_in[20];
  const float* vn_b2      = (const float*)d_in[21];
  const float* vn_ln2_g   = (const float*)d_in[22];
  const float* vn_ln2_b   = (const float*)d_in[23];
  const float* dagnn_w    = (const float*)d_in[24];
  const float* dagnn_b    = (const float*)d_in[25];
  const float* fc_w       = (const float*)d_in[26];
  const float* fc_b       = (const float*)d_in[27];
  const int*   x          = (const int*)d_in[28];
  const int*   edge_attr  = (const int*)d_in[29];
  const int*   edge_index = (const int*)d_in[30];
  const int*   batch      = (const int*)d_in[31];
  const int* src = edge_index;          // edge_index[0]
  const int* dst = edge_index + EE;     // edge_index[1]
  float* out  = (float*)d_out;
  float* mask = out + (size_t)NN * NN;

  // workspace layout
  char* ws = (char*)d_ws; size_t off = 0;
  float* h    = (float*)bump(ws, off, (size_t)NN * DD * 4);
  float* z    = (float*)bump(ws, off, (size_t)NN * DD * 4);
  float* tbuf = (float*)bump(ws, off, (size_t)NN * DD * 4);
  float* a1   = (float*)bump(ws, off, (size_t)NN * HD * 4);
  float* a2   = (float*)bump(ws, off, (size_t)NN * HD * 4);
  float* eattr= (float*)bump(ws, off, (size_t)EE * DD * 4);
  float* vn   = (float*)bump(ws, off, (size_t)GG * DD * 4);
  float* vtmp = (float*)bump(ws, off, (size_t)GG * DD * 4);
  float* va   = (float*)bump(ws, off, (size_t)GG * DD * 4);
  float* pps  = (float*)bump(ws, off, (size_t)(KHOP + 1) * NN * DD * 4);
  float* xs   = (float*)bump(ws, off, (size_t)NN * DD * 4);
  float* dis  = (float*)bump(ws, off, (size_t)NN * 4);
  int* csr_cnt= (int*)bump(ws, off, (size_t)NN * 4);
  int* csr_off= (int*)bump(ws, off, (size_t)(NN + 1) * 4);
  int* csr_e  = (int*)bump(ws, off, (size_t)EE * 4);
  const int n_win  = LL * HD * DD;
  const int n_whid = LL * 3 * HD * HD;
  const int n_wout = LL * DD * HD;
  const int n_vnw  = (LL - 1) * DD * DD;
  _Float16* w16_in  = (_Float16*)bump(ws, off, (size_t)n_win * 2);
  _Float16* w16_hid = (_Float16*)bump(ws, off, (size_t)n_whid * 2);
  _Float16* w16_out = (_Float16*)bump(ws, off, (size_t)n_wout * 2);
  _Float16* w16_vn1 = (_Float16*)bump(ws, off, (size_t)n_vnw * 2);
  _Float16* w16_vn2 = (_Float16*)bump(ws, off, (size_t)n_vnw * 2);
  _Float16* x16     = (_Float16*)bump(ws, off, (size_t)NN * HD * 2); // activation staging

  // weight conversion to f16 (once)
  k_f32_to_f16<<<(n_win + 255) / 256, 256, 0, stream>>>(w_in, w16_in, n_win);
  k_f32_to_f16<<<(n_whid + 255) / 256, 256, 0, stream>>>(w_hid, w16_hid, n_whid);
  k_f32_to_f16<<<(n_wout + 255) / 256, 256, 0, stream>>>(w_out, w16_out, n_wout);
  k_f32_to_f16<<<(n_vnw + 255) / 256, 256, 0, stream>>>(vn_w1, w16_vn1, n_vnw);
  k_f32_to_f16<<<(n_vnw + 255) / 256, 256, 0, stream>>>(vn_w2, w16_vn2, n_vnw);

  // embeddings + vn init
  k_embed_nodes<<<NN, 256, 0, stream>>>(atom_emb, x, vn0, rand_h, h);
  k_embed_edges<<<EE, 256, 0, stream>>>(bond_emb, edge_attr, rand_e, eattr);
  k_vn_init<<<GG, 256, 0, stream>>>(vn0, vn);

  // deterministic dst-CSR
  k_csr_count<<<NN / 256, 256, 0, stream>>>(dst, csr_cnt);
  k_csr_scan<<<1, 1024, 0, stream>>>(csr_cnt, csr_off);
  k_csr_fill<<<NN / 256, 256, 0, stream>>>(dst, csr_off, csr_e);

  auto gemm = [&](const float* X, const _Float16* W, const float* bias,
                  const float* res, float* Y, int M, int K, int O) {
    int nx = M * K;
    k_f32_to_f16<<<(nx + 255) / 256, 256, 0, stream>>>(X, x16, nx);
    int tiles = (M / 16) * (O / 64);
    k_gemm_wmma<<<(tiles + 7) / 8, 256, 0, stream>>>(x16, W, bias, res, Y, M, K, O);
  };

  // GNN layers
  for (int i = 0; i < LL; ++i) {
    const float* hin = h;
    if (i > 0) {
      k_ln_relu<<<NN, 256, 0, stream>>>(h, tbuf, layer_ln_g + i * DD,
                                        layer_ln_b + i * DD, DD, 1);
      hin = tbuf;
    }
    k_genconv_aggr<<<NN, 256, 0, stream>>>(hin, eattr, src, csr_off, csr_e,
                                           conv_t, i, z);
    // MLP: D -> Hd (+LN+ReLU), 3x Hd -> Hd (+LN+ReLU), Hd -> D
    gemm(z, w16_in + (size_t)i * HD * DD, b_in + i * HD, nullptr, a1, NN, DD, HD);
    k_ln_relu<<<NN, 256, 0, stream>>>(a1, a1, mlp_ln_g + (size_t)(i * 4 + 0) * HD,
                                      mlp_ln_b + (size_t)(i * 4 + 0) * HD, HD, 1);
    float* cur = a1; float* nxt = a2;
    for (int l = 0; l < 3; ++l) {
      gemm(cur, w16_hid + (size_t)(i * 3 + l) * HD * HD, b_hid + (i * 3 + l) * HD,
           nullptr, nxt, NN, HD, HD);
      k_ln_relu<<<NN, 256, 0, stream>>>(nxt, nxt, mlp_ln_g + (size_t)(i * 4 + l + 1) * HD,
                                        mlp_ln_b + (size_t)(i * 4 + l + 1) * HD, HD, 1);
      float* t_ = cur; cur = nxt; nxt = t_;
    }
    gemm(cur, w16_out + (size_t)i * DD * HD, b_out + i * DD,
         (i == 0) ? nullptr : h, h, NN, HD, DD);   // residual for layers >= 1

    if (i >= 1) {   // virtual node update
      int j = i - 1;
      k_vn_pool<<<GG, 256, 0, stream>>>(h, vn, vtmp);
      gemm(vtmp, w16_vn1 + (size_t)j * DD * DD, vn_b1 + j * DD, nullptr, va, GG, DD, DD);
      k_ln_relu<<<GG, 256, 0, stream>>>(va, va, vn_ln1_g + j * DD, vn_ln1_b + j * DD, DD, 1);
      gemm(va, w16_vn2 + (size_t)j * DD * DD, vn_b2 + j * DD, nullptr, vn, GG, DD, DD);
      k_ln_relu<<<GG, 256, 0, stream>>>(vn, vn, vn_ln2_g + j * DD, vn_ln2_b + j * DD, DD, 1);
      k_add_vn<<<NN, 256, 0, stream>>>(h, vn, batch);
    }
  }

  // final LN + vn add -> pps plane 0
  k_ln_relu<<<NN, 256, 0, stream>>>(h, tbuf, layer_ln_g, layer_ln_b, DD, 1);
  k_final_combine<<<NN, 256, 0, stream>>>(tbuf, vn, batch, pps);

  // DAGNN propagation
  k_calc_dis<<<NN / 256, 256, 0, stream>>>(csr_off, dis);
  for (int k = 1; k <= KHOP; ++k)
    k_dagnn_hop<<<NN, 256, 0, stream>>>(pps + (size_t)(k - 1) * NN * DD,
                                        pps + (size_t)k * NN * DD,
                                        dis, src, csr_off, csr_e);
  k_dagnn_score<<<NN, 256, 0, stream>>>(pps, dagnn_w, dagnn_b, xs);

  // DistMax with block-diagonal mask
  k_zero_out<<<(2 * NN * NN + 255) / 256, 256, 0, stream>>>(out);
  k_distmax<<<GG * 4, 256, 0, stream>>>(xs, fc_w, fc_b, out, mask);
}
